// CrossAttTransformer_57363583205587
// MI455X (gfx1250) — matrix operand order repaired and checked by reading, not compiled, not run on hardware
//
#include <hip/hip_runtime.h>

#define DEV static __device__ __forceinline__

typedef __attribute__((ext_vector_type(16))) _Float16 v16h;
typedef __attribute__((ext_vector_type(8)))  float    v8f;

namespace cfg {
constexpr int S = 512, Q = 32, KEYS = 128, C = 128, H = 4, KD = 32, VD = 32;
constexpr int NB = 3, PC = 16;
constexpr int HVD = H * VD;   // 128
constexpr int FC  = 256;      // F*C
constexpr int FF1 = 512;      // 2*F*C
// f16 transposed-weight arena (per block), element offsets; layout Wt[n*Kd + k]
constexpr size_t O_K_SC = 0,      O_K_SB = 16384,  O_Q_SC = 32768, O_Q_SB = 49152;
constexpr size_t O_Q_W  = 65536,  O_GATE = 81920,  O_K_W  = 98304, O_V_W  = 114688;
constexpr size_t O_AO   = 131072, O_AC   = 147456, O_F_SC = 163840, O_F_SB = 180224;
constexpr size_t O_FC   = 196608, O_FFW1 = 212992, O_FFW2 = 278528;
constexpr size_t WBLK   = 311296;  // f16 elements per block
}

struct Params {
  const float* qmask; const float* kmask; const int* q2k;
  const float* qcond; const float* kcond; const float* pair;
  const float* q_ln_g; const float* q_sc_b;
  const float* k_ln_g; const float* k_sc_b;
  const float* q_b; const float* gate_b;
  const float* ac_b; const float* f_ln_g; const float* f_sc_b; const float* fc_b;
  const float* pair_ln_g; const float* pair_w;
  const _Float16* wts;   // transposed f16 weight arena in workspace
};

DEV float warp_sum(float x) {
#pragma unroll
  for (int m = 16; m; m >>= 1) x += __shfl_xor(x, m, 32);
  return x;
}
DEV float warp_max(float x) {
#pragma unroll
  for (int m = 16; m; m >>= 1) x = fmaxf(x, __shfl_xor(x, m, 32));
  return x;
}
DEV float sigm(float x) { return 1.0f / (1.0f + __expf(-x)); }

DEV v8f vzero() {
  v8f z;
#pragma unroll
  for (int i = 0; i < 8; ++i) z[i] = 0.0f;
  return z;
}

DEV v8f wmma(v16h a, v16h b, v8f c) {
  return __builtin_amdgcn_wmma_f32_16x16x32_f16(false, a, false, b, (short)0, c, false, false);
}

// A fragment (16x32 f16) from LDS: lane m=l%16; half=l/16, k = half*16 + e (contiguous)
DEV v16h load_a_lds(const _Float16* A, int ld, int m0, int k0) {
  const int l = threadIdx.x & 31;
  const int m = m0 + (l & 15);
  const int hh = l >> 4;
  const _Float16* row = A + m * ld + k0;
  v16h f;
#pragma unroll
  for (int e = 0; e < 16; ++e) {
    int kk = hh * 16 + e;
    f[e] = row[kk];
  }
  return f;
}
DEV v16h load_a_gl(const float* A, int ld, int m0, int k0) {
  const int l = threadIdx.x & 31;
  const int m = m0 + (l & 15);
  const int hh = l >> 4;
  const float* row = A + (size_t)m * ld + k0;
  v16h f;
#pragma unroll
  for (int e = 0; e < 16; ++e) {
    int kk = hh * 16 + e;
    f[e] = (_Float16)row[kk];
  }
  return f;
}
// B fragment (32x16) from LDS, B stored [K][N]
DEV v16h load_b_lds(const _Float16* B, int ld, int k0, int n0) {
  const int l = threadIdx.x & 31;
  const int n = n0 + (l & 15);
  const int hh = l >> 4;
  v16h f;
#pragma unroll
  for (int e = 0; e < 16; ++e) f[e] = B[(k0 + hh * 16 + e) * ld + n];
  return f;
}
// B fragment from fragment-ready transposed f16 global weights: Wt[n*Kd + k].
// Per lane the 16 needed f16 are contiguous -> two global_load_b128.
DEV v16h load_b_tr(const _Float16* Bt, int Kd, int k0, int n0) {
  const int l = threadIdx.x & 31;
  const int n = n0 + (l & 15);
  const int hh = l >> 4;
  const _Float16* p = Bt + (size_t)n * Kd + k0 + hh * 16;
  union { uint4 u[2]; v16h h; } t;
  t.u[0] = *(const uint4*)(p);
  t.u[1] = *(const uint4*)(p + 8);
  return t.h;
}
// B fragment from transposed LDS storage: logical B[k][n] stored Bs[n*ld + k]
DEV v16h load_bT_lds(const _Float16* Bs, int ld, int k0, int n0) {
  const int l = threadIdx.x & 31;
  const int n = n0 + (l & 15);
  const int hh = l >> 4;
  const _Float16* row = Bs + n * ld + k0;
  v16h f;
#pragma unroll
  for (int e = 0; e < 16; ++e) f[e] = row[hh * 16 + e];
  return f;
}

// LayerNorm of a 128-wide row held as float4 per lane (use_fast_variance=False form)
DEV void ln4(float4& x) {
  float mu = warp_sum(x.x + x.y + x.z + x.w) * (1.0f / 128.0f);
  float d0 = x.x - mu, d1 = x.y - mu, d2 = x.z - mu, d3 = x.w - mu;
  float var = warp_sum(d0 * d0 + d1 * d1 + d2 * d2 + d3 * d3) * (1.0f / 128.0f);
  float rs = rsqrtf(var + 1e-5f);
  x.x = d0 * rs; x.y = d1 * rs; x.z = d2 * rs; x.w = d3 * rs;
}

// convert + transpose one weight matrix: dst[n*Kd + k] = (f16) src[k*N + n]
__global__ void wprep_kernel(const float* __restrict__ src, _Float16* __restrict__ dst,
                             int Kd, int N) {
  int i = blockIdx.x * 256 + threadIdx.x;
  if (i >= Kd * N) return;
  int k = i / N, n = i - k * N;
  dst[(size_t)n * Kd + k] = (_Float16)src[i];
}

__global__ __launch_bounds__(256) void xatt_block_kernel(int b, const float* __restrict__ actIn,
                                                         float* __restrict__ actOut, Params P) {
  using namespace cfg;
  // LDS pool (152 KB), phase-overlapped regions
  constexpr int OFF_LNQ  = 0;       // f16[32*128]  lnQ / lnA(ffw)
  constexpr int OFF_CN   = 8192;    // f16[32*128]  cnQ / cn2(ffw)
  constexpr int OFF_XQ   = 16384;   // f16[32*128]  xq  / xt(ffw)
  constexpr int OFF_Q    = 24576;   // f16[32*128]  q (pre-scaled)
  constexpr int OFF_XK   = 32768;   // f16[128*128] cnK->xk ; ffw: hid f16[32*256]
  constexpr int OFF_GATE = 65536;   // f32[32*128] gate ; phase0/1: lnK f16[128*128] 65536..98304
  constexpr int OFF_SACT = 81920;   // f32[32*128] residual act
  constexpr int OFF_KH   = 98304;   // f16[128*32]
  constexpr int OFF_VH   = 106496;  // f16[128*32]
  constexpr int OFF_LOG  = 114688;  // f32[32*128]
  constexpr int OFF_ATT  = 131072;  // f16[32*128] attn ; phase4: waG
  constexpr int OFF_WA   = 139264;  // f32[32*128]
  constexpr int LDS_BYTES = 155648;
  __shared__ __align__(16) char smem[LDS_BYTES];

  const int s = blockIdx.x;
  const int tid = threadIdx.x, wave = tid >> 5, lane = tid & 31;
  const float* actS = actIn + (size_t)s * Q * C;
  const float* qscG = P.qcond + (size_t)s * Q * C;
  const _Float16* WT = P.wts + (size_t)b * WBLK;

  _Float16* lnK  = (_Float16*)(smem + OFF_GATE);
  _Float16* xk   = (_Float16*)(smem + OFF_XK);   // cnK first, overwritten in place
  _Float16* lnQ  = (_Float16*)(smem + OFF_LNQ);
  _Float16* cnQ  = (_Float16*)(smem + OFF_CN);
  _Float16* xq   = (_Float16*)(smem + OFF_XQ);
  _Float16* qB   = (_Float16*)(smem + OFF_Q);
  float*    gate = (float*)(smem + OFF_GATE);
  float*    sAct = (float*)(smem + OFF_SACT);
  _Float16* kh   = (_Float16*)(smem + OFF_KH);
  _Float16* vh   = (_Float16*)(smem + OFF_VH);
  float*    lgts = (float*)(smem + OFF_LOG);
  _Float16* attn = (_Float16*)(smem + OFF_ATT);
  float*    wa   = (float*)(smem + OFF_WA);

  // ================= phase 0: gather keys_act, LN, cond-LN (k side) =================
  {
    _Float16* cnK = xk;
    float4 kg = *(const float4*)(P.k_ln_g + b * C + lane * 4);
    for (int r = wave; r < KEYS; r += 8) {
      int src = P.q2k[s * KEYS + r];
      float4 x = *(const float4*)(actIn + (size_t)src * C + lane * 4);
      ln4(x);
      _Float16* d = lnK + r * C + lane * 4;
      d[0] = (_Float16)x.x; d[1] = (_Float16)x.y; d[2] = (_Float16)x.z; d[3] = (_Float16)x.w;
      float4 cv = *(const float4*)(P.kcond + ((size_t)s * KEYS + r) * C + lane * 4);
      ln4(cv);
      _Float16* dc = cnK + r * C + lane * 4;
      dc[0] = (_Float16)(cv.x * kg.x); dc[1] = (_Float16)(cv.y * kg.y);
      dc[2] = (_Float16)(cv.z * kg.z); dc[3] = (_Float16)(cv.w * kg.w);
    }
  }
  __syncthreads();

  // ================= phase 1: xk = sigmoid(cnK@sc_w + sc_b)*lnK + cnK@sb_w =================
  {
    const float* Bsc = P.k_sc_b + b * C;
    const int mt = wave;  // 8 row tiles, one per wave
    v16h a4[4];
#pragma unroll
    for (int kk = 0; kk < 4; ++kk) a4[kk] = load_a_lds(xk, C, mt * 16, kk * 32);
    v8f ac1[8], ac2[8];
#pragma unroll
    for (int i = 0; i < 8; ++i) { ac1[i] = vzero(); ac2[i] = vzero(); }
#pragma unroll
    for (int nt = 0; nt < 8; ++nt)
#pragma unroll
      for (int kk = 0; kk < 4; ++kk) {
        ac1[nt] = wmma(a4[kk], load_b_tr(WT + O_K_SC, C, kk * 32, nt * 16), ac1[nt]);
        ac2[nt] = wmma(a4[kk], load_b_tr(WT + O_K_SB, C, kk * 32, nt * 16), ac2[nt]);
      }
#pragma unroll
    for (int nt = 0; nt < 8; ++nt) {
      int n = nt * 16 + (lane & 15);
      float sb = Bsc[n];
#pragma unroll
      for (int v = 0; v < 8; ++v) {
        int m = mt * 16 + v + 8 * (lane >> 4);
        float sg = sigm(ac1[nt][v] + sb);
        xk[m * C + n] = (_Float16)(sg * (float)lnK[m * C + n] + ac2[nt][v]);
      }
    }
  }
  __syncthreads();

  // ================= phase 2: q side — lnQ/cnQ, xq, q(+scale), gate =================
  {
    float4 qg = *(const float4*)(P.q_ln_g + b * C + lane * 4);
    for (int r = wave; r < Q; r += 8) {
      float4 x = *(const float4*)(actS + r * C + lane * 4);
      ln4(x);
      _Float16* d = lnQ + r * C + lane * 4;
      d[0] = (_Float16)x.x; d[1] = (_Float16)x.y; d[2] = (_Float16)x.z; d[3] = (_Float16)x.w;
      float4 cv = *(const float4*)(qscG + r * C + lane * 4);
      ln4(cv);
      _Float16* dc = cnQ + r * C + lane * 4;
      dc[0] = (_Float16)(cv.x * qg.x); dc[1] = (_Float16)(cv.y * qg.y);
      dc[2] = (_Float16)(cv.z * qg.z); dc[3] = (_Float16)(cv.w * qg.w);
    }
  }
  __syncthreads();
  {
    const float* Bsc = P.q_sc_b + b * C;
    for (int u = wave * 2; u < wave * 2 + 2; ++u) {
      int mt = u >> 3, nt = u & 7;
      v8f a1 = vzero(), a2 = vzero();
#pragma unroll
      for (int kk = 0; kk < 4; ++kk) {
        v16h a = load_a_lds(cnQ, C, mt * 16, kk * 32);
        a1 = wmma(a, load_b_tr(WT + O_Q_SC, C, kk * 32, nt * 16), a1);
        a2 = wmma(a, load_b_tr(WT + O_Q_SB, C, kk * 32, nt * 16), a2);
      }
      int n = nt * 16 + (lane & 15);
      float sb = Bsc[n];
#pragma unroll
      for (int v = 0; v < 8; ++v) {
        int m = mt * 16 + v + 8 * (lane >> 4);
        float sg = sigm(a1[v] + sb);
        xq[m * C + n] = (_Float16)(sg * (float)lnQ[m * C + n] + a2[v]);
      }
    }
  }
  __syncthreads();
  {
    const float scale = 0.17677669529663687f;  // KD^-0.5
    for (int u = wave * 2; u < wave * 2 + 2; ++u) {
      int mt = u >> 3, nt = u & 7;
      v8f aq = vzero(), ag = vzero();
#pragma unroll
      for (int kk = 0; kk < 4; ++kk) {
        v16h a = load_a_lds(xq, C, mt * 16, kk * 32);
        aq = wmma(a, load_b_tr(WT + O_Q_W, C, kk * 32, nt * 16), aq);
        ag = wmma(a, load_b_tr(WT + O_GATE, C, kk * 32, nt * 16), ag);
      }
      int n = nt * 16 + (lane & 15);
      float qb0 = P.q_b[b * HVD + n];
      float gb0 = P.gate_b[b * HVD + n];
#pragma unroll
      for (int v = 0; v < 8; ++v) {
        int m = mt * 16 + v + 8 * (lane >> 4);
        qB[m * HVD + n] = (_Float16)((aq[v] + qb0) * scale);
        gate[m * HVD + n] = sigm(ag[v] + gb0);  // overwrites lnK region (dead)
      }
    }
  }
  __syncthreads();

  // ================= phase 3: attention, per head =================
  for (int h = 0; h < H; ++h) {
    // (a) k_h, v_h projections [128 x 32]
    for (int u = wave * 2; u < wave * 2 + 2; ++u) {
      int mt = u >> 1, nt = u & 1;
      v8f acK = vzero(), acV = vzero();
#pragma unroll
      for (int kk = 0; kk < 4; ++kk) {
        v16h a = load_a_lds(xk, C, mt * 16, kk * 32);
        acK = wmma(a, load_b_tr(WT + O_K_W, C, kk * 32, h * KD + nt * 16), acK);
        acV = wmma(a, load_b_tr(WT + O_V_W, C, kk * 32, h * VD + nt * 16), acV);
      }
      int n = nt * 16 + (lane & 15);
#pragma unroll
      for (int v = 0; v < 8; ++v) {
        int m = mt * 16 + v + 8 * (lane >> 4);
        kh[m * KD + n] = (_Float16)acK[v];
        vh[m * VD + n] = (_Float16)acV[v];
      }
    }
    __syncthreads();
    // (b) logits = q_h k_h^T + bias + pair-logits
    float gw[PC];
#pragma unroll
    for (int p = 0; p < PC; ++p) gw[p] = P.pair_ln_g[p] * P.pair_w[(p * NB + b) * H + h];
    for (int u = wave * 2; u < wave * 2 + 2; ++u) {
      int mt = u >> 3, nt = u & 7;
      v8f ac = vzero();
      v16h a = load_a_lds(qB, HVD, mt * 16, h * KD);
      v16h bf = load_bT_lds(kh, KD, 0, nt * 16);
      ac = wmma(a, bf, ac);
      int n = nt * 16 + (lane & 15);
      float km1 = P.kmask[s * KEYS + n] - 1.0f;
#pragma unroll
      for (int v = 0; v < 8; ++v) {
        int m = mt * 16 + v + 8 * (lane >> 4);
        float qm1 = P.qmask[s * Q + m] - 1.0f;
        const float* pc = P.pair + ((size_t)(s * Q + m) * KEYS + n) * PC;
        if (v < 7) __builtin_prefetch(pc + (size_t)KEYS * PC, 0, 0);  // next row's pair data
        float pcv[16];
        *(float4*)(pcv + 0)  = *(const float4*)(pc + 0);
        *(float4*)(pcv + 4)  = *(const float4*)(pc + 4);
        *(float4*)(pcv + 8)  = *(const float4*)(pc + 8);
        *(float4*)(pcv + 12) = *(const float4*)(pc + 12);
        float mu = 0.0f;
#pragma unroll
        for (int p = 0; p < 16; ++p) mu += pcv[p];
        mu *= (1.0f / 16.0f);
        float var = 0.0f, pl = 0.0f;
#pragma unroll
        for (int p = 0; p < 16; ++p) {
          float d = pcv[p] - mu;
          var += d * d;
          pl += d * gw[p];
        }
        pl *= rsqrtf(var * (1.0f / 16.0f) + 1e-5f);
        lgts[m * KEYS + n] = ac[v] + 1e9f * qm1 * km1 + pl;
      }
    }
    __syncthreads();
    // (c) softmax over keys
    for (int r = wave; r < Q; r += 8) {
      float4 x = *(float4*)(lgts + r * KEYS + lane * 4);
      float mx = warp_max(fmaxf(fmaxf(x.x, x.y), fmaxf(x.z, x.w)));
      float e0 = __expf(x.x - mx), e1 = __expf(x.y - mx);
      float e2 = __expf(x.z - mx), e3 = __expf(x.w - mx);
      float inv = 1.0f / warp_sum(e0 + e1 + e2 + e3);
      _Float16* d = attn + r * KEYS + lane * 4;
      d[0] = (_Float16)(e0 * inv); d[1] = (_Float16)(e1 * inv);
      d[2] = (_Float16)(e2 * inv); d[3] = (_Float16)(e3 * inv);
    }
    __syncthreads();
    // (d) wa_h = attn @ v_h [32 x 32]
    if (wave < 4) {
      int mt = wave >> 1, nt = wave & 1;
      v8f ac = vzero();
#pragma unroll
      for (int kk = 0; kk < 4; ++kk)
        ac = wmma(load_a_lds(attn, KEYS, mt * 16, kk * 32),
                  load_b_lds(vh, VD, kk * 32, nt * 16), ac);
      int n = nt * 16 + (lane & 15);
#pragma unroll
      for (int v = 0; v < 8; ++v) {
        int m = mt * 16 + v + 8 * (lane >> 4);
        wa[m * HVD + h * VD + n] = ac[v];
      }
    }
    __syncthreads();
  }

  // ================= phase 4: out = (wa*gate)@ao_w * sigmoid(qsc@ac_w+ac_b); act += out ===
  {
    _Float16* waG = attn;  // reuse OFF_ATT (attn dead)
    for (int i = tid; i < Q * C; i += 256) waG[i] = (_Float16)(wa[i] * gate[i]);
    {
      // CDNA5 async DMA: copy this subset's act rows (16 KB) straight into LDS,
      // bypassing VGPRs; tracked by ASYNCcnt.
      unsigned int laddr = (unsigned int)(uintptr_t)(sAct + tid * 4);
      const float* gaddr = actS + tid * 4;
#pragma unroll
      for (int it = 0; it < 4; ++it) {
        asm volatile("global_load_async_to_lds_b128 %0, %1, off"
                     :: "v"(laddr + it * 4096u), "v"(gaddr + it * 1024) : "memory");
      }
      asm volatile("s_wait_asynccnt 0x0" ::: "memory");
    }
    __syncthreads();
    for (int u = wave * 2; u < wave * 2 + 2; ++u) {
      int mt = u >> 3, nt = u & 7;
      v8f aO = vzero(), aC = vzero();
#pragma unroll
      for (int kk = 0; kk < 4; ++kk) {
        aO = wmma(load_a_lds(waG, HVD, mt * 16, kk * 32),
                  load_b_tr(WT + O_AO, HVD, kk * 32, nt * 16), aO);
        aC = wmma(load_a_gl(qscG, C, mt * 16, kk * 32),
                  load_b_tr(WT + O_AC, C, kk * 32, nt * 16), aC);
      }
      int n = nt * 16 + (lane & 15);
      float cb = P.ac_b[b * C + n];
#pragma unroll
      for (int v = 0; v < 8; ++v) {
        int m = mt * 16 + v + 8 * (lane >> 4);
        sAct[m * C + n] += aO[v] * sigm(aC[v] + cb);
      }
    }
  }
  __syncthreads();

  // ================= phase 5: GLU transition =================
  {
    _Float16* lnA = lnQ;  // reuse
    _Float16* cn2 = cnQ;
    _Float16* xt  = xq;
    _Float16* hid = (_Float16*)(smem + OFF_XK);  // xk dead
    float4 fg = *(const float4*)(P.f_ln_g + b * C + lane * 4);
    for (int r = wave; r < Q; r += 8) {
      float4 x = *(float4*)(sAct + r * C + lane * 4);
      ln4(x);
      _Float16* d = lnA + r * C + lane * 4;
      d[0] = (_Float16)x.x; d[1] = (_Float16)x.y; d[2] = (_Float16)x.z; d[3] = (_Float16)x.w;
      float4 cv = *(const float4*)(qscG + r * C + lane * 4);
      ln4(cv);
      _Float16* dc = cn2 + r * C + lane * 4;
      dc[0] = (_Float16)(cv.x * fg.x); dc[1] = (_Float16)(cv.y * fg.y);
      dc[2] = (_Float16)(cv.z * fg.z); dc[3] = (_Float16)(cv.w * fg.w);
    }
    __syncthreads();
    {
      const float* Bsc = P.f_sc_b + b * C;
      for (int u = wave * 2; u < wave * 2 + 2; ++u) {
        int mt = u >> 3, nt = u & 7;
        v8f a1 = vzero(), a2 = vzero();
#pragma unroll
        for (int kk = 0; kk < 4; ++kk) {
          v16h a = load_a_lds(cn2, C, mt * 16, kk * 32);
          a1 = wmma(a, load_b_tr(WT + O_F_SC, C, kk * 32, nt * 16), a1);
          a2 = wmma(a, load_b_tr(WT + O_F_SB, C, kk * 32, nt * 16), a2);
        }
        int n = nt * 16 + (lane & 15);
        float sb = Bsc[n];
#pragma unroll
        for (int v = 0; v < 8; ++v) {
          int m = mt * 16 + v + 8 * (lane >> 4);
          float sg = sigm(a1[v] + sb);
          xt[m * C + n] = (_Float16)(sg * (float)lnA[m * C + n] + a2[v]);
        }
      }
    }
    __syncthreads();
    {  // ffw1: hid = swish(xt@W1[:, :256]) * (xt@W1[:, 256:])
      for (int u = wave * 4; u < wave * 4 + 4; ++u) {
        int mt = u >> 4, j = u & 15;
        v8f aA = vzero(), aB = vzero();
#pragma unroll
        for (int kk = 0; kk < 4; ++kk) {
          v16h a = load_a_lds(xt, C, mt * 16, kk * 32);
          aA = wmma(a, load_b_tr(WT + O_FFW1, C, kk * 32, j * 16), aA);
          aB = wmma(a, load_b_tr(WT + O_FFW1, C, kk * 32, FC + j * 16), aB);
        }
        int n = j * 16 + (lane & 15);
#pragma unroll
        for (int v = 0; v < 8; ++v) {
          int m = mt * 16 + v + 8 * (lane >> 4);
          float av = aA[v];
          hid[m * FC + n] = (_Float16)(av * sigm(av) * aB[v]);
        }
      }
    }
    __syncthreads();
    {  // ffw2 + fc gate + residual -> actOut
      for (int u = wave * 2; u < wave * 2 + 2; ++u) {
        int mt = u >> 3, nt = u & 7;
        v8f aT = vzero(), aF = vzero();
#pragma unroll
        for (int kk = 0; kk < 8; ++kk)
          aT = wmma(load_a_lds(hid, FC, mt * 16, kk * 32),
                    load_b_tr(WT + O_FFW2, FC, kk * 32, nt * 16), aT);
#pragma unroll
        for (int kk = 0; kk < 4; ++kk)
          aF = wmma(load_a_gl(qscG, C, mt * 16, kk * 32),
                    load_b_tr(WT + O_FC, C, kk * 32, nt * 16), aF);
        int n = nt * 16 + (lane & 15);
        float fb = P.fc_b[b * C + n];
#pragma unroll
        for (int v = 0; v < 8; ++v) {
          int m = mt * 16 + v + 8 * (lane >> 4);
          float sg = sigm(aF[v] + fb);
          actOut[((size_t)s * Q + m) * C + n] = sAct[m * C + n] + aT[v] * sg;
        }
      }
    }
  }
}

extern "C" void kernel_launch(void* const* d_in, const int* in_sizes, int n_in,
                              void* d_out, int out_size, void* d_ws, size_t ws_size,
                              hipStream_t stream) {
  (void)in_sizes; (void)n_in; (void)out_size; (void)ws_size;
  using namespace cfg;
  const float* act0 = (const float*)d_in[0];
  Params P;
  P.qmask     = (const float*)d_in[1];
  P.q2k       = (const int*)d_in[2];
  P.kmask     = (const float*)d_in[3];
  P.qcond     = (const float*)d_in[4];
  P.kcond     = (const float*)d_in[5];
  P.pair      = (const float*)d_in[6];
  P.q_ln_g    = (const float*)d_in[7];
  P.q_sc_b    = (const float*)d_in[9];
  P.k_ln_g    = (const float*)d_in[11];
  P.k_sc_b    = (const float*)d_in[13];
  P.q_b       = (const float*)d_in[16];
  P.gate_b    = (const float*)d_in[20];
  P.ac_b      = (const float*)d_in[23];
  P.f_ln_g    = (const float*)d_in[24];
  P.f_sc_b    = (const float*)d_in[26];
  P.fc_b      = (const float*)d_in[31];
  P.pair_ln_g = (const float*)d_in[32];
  P.pair_w    = (const float*)d_in[33];

  float* wsA = (float*)d_ws;
  float* wsB = wsA + (size_t)S * Q * C;
  _Float16* wts = (_Float16*)(wsB + (size_t)S * Q * C);
  P.wts = wts;

  // ---- weight prep: f32 -> f16, transposed to fragment-ready Wt[n][k] ----
  struct WD { int idx; int Kd, N; size_t off; };
  const WD wd[15] = {
    {12, 128, 128, O_K_SC}, {14, 128, 128, O_K_SB}, {8, 128, 128, O_Q_SC},
    {10, 128, 128, O_Q_SB}, {15, 128, 128, O_Q_W},  {19, 128, 128, O_GATE},
    {17, 128, 128, O_K_W},  {18, 128, 128, O_V_W},  {21, 128, 128, O_AO},
    {22, 128, 128, O_AC},   {25, 128, 128, O_F_SC}, {27, 128, 128, O_F_SB},
    {30, 128, 128, O_FC},   {28, 128, 512, O_FFW1}, {29, 256, 128, O_FFW2},
  };
  for (int b = 0; b < NB; ++b)
    for (int i = 0; i < 15; ++i) {
      int total = wd[i].Kd * wd[i].N;
      wprep_kernel<<<(total + 255) / 256, 256, 0, stream>>>(
          (const float*)d_in[wd[i].idx] + (size_t)b * total,
          wts + (size_t)b * WBLK + wd[i].off, wd[i].Kd, wd[i].N);
    }

  // ---- scan over NB=3 blocks; kernel boundaries give the global sync the
  // cross-subset key gather requires; act ping-pongs through workspace ----
  dim3 grid(S), blk(256);
  xatt_block_kernel<<<grid, blk, 0, stream>>>(0, act0, wsA, P);
  xatt_block_kernel<<<grid, blk, 0, stream>>>(1, wsA, wsB, P);
  xatt_block_kernel<<<grid, blk, 0, stream>>>(2, wsB, (float*)d_out, P);
}
